// MultiHeadAttention_64029372449075
// MI455X (gfx1250) — matrix operand order repaired and checked by reading, not compiled, not run on hardware
//
#include <hip/hip_runtime.h>
#include <hip/hip_bf16.h>

#define E_DIM  2048
#define HEADS  16
#define DHEAD  128
#define LSEQ   2048
#define NBATCH 4
#define MTOT   (NBATCH * LSEQ) /* 8192 */

typedef __attribute__((ext_vector_type(16))) __bf16       bf16x16;
typedef __attribute__((ext_vector_type(8)))  float        f32x8;
typedef __attribute__((ext_vector_type(4)))  float        f32x4;
typedef __attribute__((ext_vector_type(4)))  unsigned int u32x4;

union Frag {
    bf16x16      bf;
    u32x4        q[2];
    unsigned int u[8];
};

__device__ __forceinline__ unsigned short f2bf(float f) {
    unsigned int u = __float_as_uint(f);
    unsigned int r = 0x7FFFu + ((u >> 16) & 1u); // round-to-nearest-even
    return (unsigned short)((u + r) >> 16);
}
__device__ __forceinline__ unsigned int pack2(float lo, float hi) {
    return (unsigned int)f2bf(lo) | ((unsigned int)f2bf(hi) << 16);
}
__device__ __forceinline__ f32x8 wmma_bf16(const Frag& a, const Frag& b, f32x8 c) {
    // D = A(16x32 bf16) * B(32x16 bf16) + C(16x16 f32)
    return __builtin_amdgcn_wmma_f32_16x16x32_bf16(
        false, a.bf, false, b.bf, (short)0, c, false, false);
}

// -------------------------------------------------------------------------
// GEMM: C[M=8192, N=2048] = X[M, K=2048] @ W[N, K]^T
// mode 0: store bf16 as q/k layout [n][h][l][d]
// mode 2: store bf16 as v^T layout [n][h][d][l]
// mode 3: store f32 (X @ Wo^T + bias) to d_out
// -------------------------------------------------------------------------
template <typename XT>
__global__ __launch_bounds__(256) void gemm_xwt(
    const XT* __restrict__ X, const float* __restrict__ W,
    const float* __restrict__ bias,
    unsigned short* __restrict__ outb, float* __restrict__ outf, int mode)
{
    __shared__ unsigned short As[128 * 40]; // 128 x 32 bf16, stride 40 (pad)
    __shared__ unsigned short Bs[64 * 40];  //  64 x 32 bf16, stride 40

    const int tid  = threadIdx.x;
    const int lane = tid & 31;
    const int wave = tid >> 5;
    const int wm   = wave & 3;   // 4 row-waves * 32 rows = 128
    const int wn   = wave >> 2;  // 2 col-waves * 32 cols = 64
    const int m0   = blockIdx.x * 128;
    const int n0   = blockIdx.y * 64;

    f32x8 acc[2][2] = {};

    const int arow = tid >> 1;           // 0..127
    const int ac0  = (tid & 1) * 16;     // 0 or 16
    const int brow = tid & 63;           // 0..63
    const int bc0  = (tid >> 6) * 8;     // 0,8,16,24

    const int kb   = (lane >> 4) * 16;   // A-frag K base per half-wave
    const int krow = (lane >> 4) * 16;   // B-frag K base per half-wave
    const int l15  = lane & 15;

    for (int kk = 0; kk < E_DIM; kk += 32) {
        // ---- stage A tile (128 x 32) into LDS as bf16 ----
        if constexpr (sizeof(XT) == 4) {
            const float* xp = (const float*)X + (size_t)(m0 + arow) * E_DIM + kk + ac0;
            f32x4 x0 = *(const f32x4*)(xp + 0);
            f32x4 x1 = *(const f32x4*)(xp + 4);
            f32x4 x2 = *(const f32x4*)(xp + 8);
            f32x4 x3 = *(const f32x4*)(xp + 12);
            u32x4 p0 = { pack2(x0[0], x0[1]), pack2(x0[2], x0[3]),
                         pack2(x1[0], x1[1]), pack2(x1[2], x1[3]) };
            u32x4 p1 = { pack2(x2[0], x2[1]), pack2(x2[2], x2[3]),
                         pack2(x3[0], x3[1]), pack2(x3[2], x3[3]) };
            *(u32x4*)&As[arow * 40 + ac0]     = p0;
            *(u32x4*)&As[arow * 40 + ac0 + 8] = p1;
        } else {
            const unsigned short* xp =
                (const unsigned short*)X + (size_t)(m0 + arow) * E_DIM + kk + ac0;
            *(u32x4*)&As[arow * 40 + ac0]     = *(const u32x4*)(xp + 0);
            *(u32x4*)&As[arow * 40 + ac0 + 8] = *(const u32x4*)(xp + 8);
        }
        // ---- stage B tile = W rows n0..n0+63, cols kk..kk+31, as bf16 ----
        {
            const float* wp = W + (size_t)(n0 + brow) * E_DIM + kk + bc0;
            f32x4 w0 = *(const f32x4*)(wp + 0);
            f32x4 w1 = *(const f32x4*)(wp + 4);
            u32x4 p  = { pack2(w0[0], w0[1]), pack2(w0[2], w0[3]),
                         pack2(w1[0], w1[1]), pack2(w1[2], w1[3]) };
            *(u32x4*)&Bs[brow * 40 + bc0] = p;
        }
        __syncthreads();

        Frag a[2], b[2];
#pragma unroll
        for (int im = 0; im < 2; ++im) {
            int mr = wm * 32 + im * 16 + l15;
            a[im].q[0] = *(const u32x4*)&As[mr * 40 + kb];
            a[im].q[1] = *(const u32x4*)&As[mr * 40 + kb + 8];
        }
#pragma unroll
        for (int in = 0; in < 2; ++in) {
            int nc = wn * 32 + in * 16 + l15;
            b[in].q[0] = *(const u32x4*)&Bs[nc * 40 + krow];
            b[in].q[1] = *(const u32x4*)&Bs[nc * 40 + krow + 8];
        }
#pragma unroll
        for (int im = 0; im < 2; ++im)
#pragma unroll
            for (int in = 0; in < 2; ++in)
                acc[im][in] = wmma_bf16(a[im], b[in], acc[im][in]);
        __syncthreads();
    }

    // ---- epilogue: scatter per C-layout (lane=N col, VGPR r -> row r+8*(lane>>4))
#pragma unroll
    for (int im = 0; im < 2; ++im) {
#pragma unroll
        for (int in = 0; in < 2; ++in) {
            int mbase = m0 + wm * 32 + im * 16 + 8 * (lane >> 4);
            int nglob = n0 + wn * 32 + in * 16 + l15;
#pragma unroll
            for (int r = 0; r < 8; ++r) {
                int   m = mbase + r;
                float v = acc[im][in][r];
                if (mode == 3) {
                    outf[(size_t)m * E_DIM + nglob] = v + bias[nglob];
                } else {
                    int nb = m >> 11, l = m & (LSEQ - 1);
                    int hh = nglob >> 7, d = nglob & (DHEAD - 1);
                    size_t idx = (mode == 2)
                        ? ((size_t)(nb * HEADS + hh) * DHEAD + d) * LSEQ + l   // v^T
                        : ((size_t)(nb * HEADS + hh) * LSEQ + l) * DHEAD + d;  // q,k
                    outb[idx] = f2bf(v);
                }
            }
        }
    }
}

// -------------------------------------------------------------------------
// Flash attention: one block per (n, h, 64 q-rows); 4 waves x 16 q-rows.
// Per 32-key tile: load all K+V fragments up front, 8 QK WMMA, softmax
// (V-load latency hides under the exp/shuffle VALU work), 8 PV WMMA.
// -------------------------------------------------------------------------
__global__ __launch_bounds__(128) void flash_attn(
    const unsigned short* __restrict__ qm, const unsigned short* __restrict__ km,
    const unsigned short* __restrict__ vtm, const int* __restrict__ mask,
    unsigned short* __restrict__ att)
{
    __shared__ unsigned short Ps[4 * 16 * 40]; // per-wave 16x32 bf16 P tile (pad 40)

    const int lane  = threadIdx.x & 31;
    const int wave  = threadIdx.x >> 5;
    const int n     = blockIdx.z, h = blockIdx.y;
    const int q0    = blockIdx.x * 64 + wave * 16;
    const int wbase = wave * 16 * 40;
    const int l15   = lane & 15;
    const int kb    = (lane >> 4) * 16;
    const int k16   = (lane >> 4) * 16;
    const float SCALE = 0.022097086912079608f; // E^-0.5, E=2048

    const size_t head = (size_t)(n * HEADS + h);

    // preload Q (16 x 128) as 4 A-fragments, straight from global (d-contiguous)
    const unsigned short* qrow = qm + (head * LSEQ + q0 + l15) * DHEAD;
    Frag qf[4];
#pragma unroll
    for (int dc = 0; dc < 4; ++dc) {
        qf[dc].q[0] = *(const u32x4*)(qrow + dc * 32 + kb);
        qf[dc].q[1] = *(const u32x4*)(qrow + dc * 32 + kb + 8);
    }

    f32x8 o[8] = {};
    float mrow[8], lrow[8];
#pragma unroll
    for (int r = 0; r < 8; ++r) { mrow[r] = -1e30f; lrow[r] = 0.f; }

    const int* mptr = mask + n * LSEQ;

    for (int kt = 0; kt < LSEQ; kt += 32) {
        const int mv0 = mptr[kt + l15];
        const int mv1 = mptr[kt + 16 + l15];

        // ---- fragment loads for the whole tile, issued before any compute ----
        const unsigned short* kp0 = km + (head * LSEQ + kt + l15) * DHEAD + k16;
        const unsigned short* kp1 = kp0 + 16 * DHEAD;
        const unsigned short* vp  = vtm + (head * DHEAD + l15) * LSEQ + kt + k16;
        __builtin_prefetch(kp0 + 32 * DHEAD, 0, 1);
        __builtin_prefetch(vp + 32, 0, 1);

        Frag kf0[4], kf1[4];
#pragma unroll
        for (int dc = 0; dc < 4; ++dc) {
            kf0[dc].q[0] = *(const u32x4*)(kp0 + dc * 32);
            kf0[dc].q[1] = *(const u32x4*)(kp0 + dc * 32 + 8);
            kf1[dc].q[0] = *(const u32x4*)(kp1 + dc * 32);
            kf1[dc].q[1] = *(const u32x4*)(kp1 + dc * 32 + 8);
        }
        Frag vf[8];
#pragma unroll
        for (int dt = 0; dt < 8; ++dt) {
            vf[dt].q[0] = *(const u32x4*)(vp + (size_t)dt * 16 * LSEQ);
            vf[dt].q[1] = *(const u32x4*)(vp + (size_t)dt * 16 * LSEQ + 8);
        }

        // ---- S = Q K^T : two 16x16 score tiles, contraction over d (4 chunks)
        f32x8 s0 = {}, s1 = {};
#pragma unroll
        for (int dc = 0; dc < 4; ++dc) {
            s0 = wmma_bf16(qf[dc], kf0[dc], s0);
            s1 = wmma_bf16(qf[dc], kf1[dc], s1);
        }

        // ---- online softmax; row stats reduced across the 16-lane N-groups
        float alph[8];
#pragma unroll
        for (int r = 0; r < 8; ++r) {
            float a = mv0 ? s0[r] * SCALE : -1e30f;
            float b = mv1 ? s1[r] * SCALE : -1e30f;
            float t = fmaxf(a, b);
            t = fmaxf(t, __shfl_xor(t, 1));
            t = fmaxf(t, __shfl_xor(t, 2));
            t = fmaxf(t, __shfl_xor(t, 4));
            t = fmaxf(t, __shfl_xor(t, 8));
            float mn = fmaxf(mrow[r], t);
            float al = __expf(mrow[r] - mn);
            float p0 = __expf(a - mn);
            float p1 = __expf(b - mn);
            float rs = p0 + p1;
            rs += __shfl_xor(rs, 1);
            rs += __shfl_xor(rs, 2);
            rs += __shfl_xor(rs, 4);
            rs += __shfl_xor(rs, 8);
            lrow[r] = lrow[r] * al + rs;
            mrow[r] = mn;
            alph[r] = al;
            // D-layout -> [q][k] transpose staging in per-wave LDS tile
            int prow = r + 8 * (lane >> 4);
            Ps[wbase + prow * 40 + l15]      = f2bf(p0);
            Ps[wbase + prow * 40 + 16 + l15] = f2bf(p1);
        }
#pragma unroll
        for (int dt = 0; dt < 8; ++dt)
#pragma unroll
            for (int r = 0; r < 8; ++r)
                o[dt][r] *= alph[r];

        // intra-wave LDS transpose hazard: wait for DS stores before A-frag loads
        asm volatile("s_wait_dscnt 0" ::: "memory");

        Frag pf; // P as A-fragment (16 q x 32 k)
        pf.q[0] = *(const u32x4*)&Ps[wbase + l15 * 40 + kb];
    pf.q[1] = *(const u32x4*)&Ps[wbase + l15 * 40 + kb + 8];

        // ---- O += P V
#pragma unroll
        for (int dt = 0; dt < 8; ++dt)
            o[dt] = wmma_bf16(pf, vf[dt], o[dt]);
    }

    // ---- normalize and store bf16 attention output as (N, L, E)
#pragma unroll
    for (int dt = 0; dt < 8; ++dt) {
#pragma unroll
        for (int r = 0; r < 8; ++r) {
            int   m = q0 + r + 8 * (lane >> 4);
            int   e = h * DHEAD + dt * 16 + l15;
            float v = o[dt][r] / (lrow[r] + 1e-37f);
            att[((size_t)n * LSEQ + m) * E_DIM + e] = f2bf(v);
        }
    }
}

// -------------------------------------------------------------------------
// Host launcher
// in: 0 query, 1 keys, 2 values, 3 mask(int32), 4 Wq, 5 Wk, 6 Wv, 7 Wo, 8 bo
// -------------------------------------------------------------------------
extern "C" void kernel_launch(void* const* d_in, const int* in_sizes, int n_in,
                              void* d_out, int out_size, void* d_ws, size_t ws_size,
                              hipStream_t stream)
{
    const size_t ELEMS = (size_t)MTOT * E_DIM;       // 16,777,216 per tensor
    const size_t NEED  = 4 * ELEMS * sizeof(unsigned short);
    if (ws_size < NEED) return;

    unsigned short* qb   = (unsigned short*)d_ws;
    unsigned short* kbuf = qb + ELEMS;
    unsigned short* vtb  = kbuf + ELEMS;
    unsigned short* attb = vtb + ELEMS;

    const dim3 gg(MTOT / 128, E_DIM / 64, 1); // 64 x 32 blocks
    const dim3 gb(256, 1, 1);

    gemm_xwt<float><<<gg, gb, 0, stream>>>(
        (const float*)d_in[0], (const float*)d_in[4], nullptr, qb, nullptr, 0);
    gemm_xwt<float><<<gg, gb, 0, stream>>>(
        (const float*)d_in[1], (const float*)d_in[5], nullptr, kbuf, nullptr, 0);
    gemm_xwt<float><<<gg, gb, 0, stream>>>(
        (const float*)d_in[2], (const float*)d_in[6], nullptr, vtb, nullptr, 2);

    flash_attn<<<dim3(LSEQ / 64, HEADS, NBATCH), dim3(128, 1, 1), 0, stream>>>(
        qb, kbuf, vtb, (const int*)d_in[3], attb);

    gemm_xwt<unsigned short><<<gg, gb, 0, stream>>>(
        attb, (const float*)d_in[7], (const float*)d_in[8],
        nullptr, (float*)d_out, 3);
}
